// CarS_764504179132
// MI455X (gfx1250) — compile-verified, matrix operand order checked by reading, not run on hardware
//
#include <hip/hip_runtime.h>
#include <hip/hip_bf16.h>

// Sparse 3D conv net forward for MI455X (gfx1250).
// Strategy: fused gather -> WMMA GEMM -> atomic scatter per rulebook offset.
//  * 96x96 convs: v_wmma_f32_16x16x32_f16, K=96 in 3 steps, f32 accum.
//  * 4->96 input conv: v_wmma_f32_16x16x4_f32 (exact K=4 fit, full f32).
//  * Conv biases dropped (cancel exactly inside BN).
//  * Wave-uniform early exit for all-pad rulebook tiles (rulebooks are padded
//    to the max per-offset count; most offsets are nearly empty on sparse data).
//  * Everything stays L2-resident (~30MB working set vs 192MB L2).
// Param input ordering assumption: jax pytree flatten (sorted dict keys).

typedef __attribute__((ext_vector_type(16))) _Float16 v16h;
typedef __attribute__((ext_vector_type(8)))  float    v8f;
typedef __attribute__((ext_vector_type(2)))  float    v2f;

#define CH   96
#define KOFF 27

__device__ __forceinline__ int kmap16(int j, int hi) {
  // element j (0..15) of a v16h A/B fragment, half hi -> K index within a 32-block
  // (per ISA 16-bit A-matrix 16x32 layout: VGPR0-3 = K 0..7 (+8*hi), VGPR4-7 = K 16..23 (+8*hi))
  return (j < 8) ? (hi * 8 + j) : (16 + hi * 8 + (j - 8));
}

__global__ void fill_zero_f32(float* __restrict__ p, long n) {
  long i = (long)blockIdx.x * blockDim.x + threadIdx.x;
  long stride = (long)gridDim.x * blockDim.x;
  for (; i < n; i += stride) p[i] = 0.0f;
}

__global__ void copy_f32(const float* __restrict__ s, float* __restrict__ d, int n) {
  int i = blockIdx.x * blockDim.x + threadIdx.x;
  if (i < n) d[i] = s[i];
}

// Repack a (27,96,96) f32 weight into per-(ko,kb,nb) WMMA B fragments (f16).
// Layout: B[ko][kb][nb][lane 0..31][j 0..15], contiguous -> one v16h load per lane.
__global__ void repack_w96(const float* __restrict__ W, _Float16* __restrict__ B) {
  int t = blockIdx.x * blockDim.x + threadIdx.x;
  const int total = KOFF * 3 * 6 * 32 * 16;
  if (t >= total) return;
  int j    = t & 15;
  int lane = (t >> 4) & 31;
  int nb   = (t >> 9) % 6;       // /512 % 6
  int kb   = (t / 3072) % 3;
  int ko   = t / 9216;
  int hi = lane >> 4, lr = lane & 15;
  int ci = kb * 32 + kmap16(j, hi);     // B "row" = K index (input channel)
  int co = nb * 16 + lr;                // B column = output channel
  B[t] = (_Float16)W[((long)ko * CH + ci) * CH + co];
}

// Repack (27,4,96) f32 input weight into f32 B fragments for 16x16x4 WMMA.
// Layout: B[ko][nb][lane][i 0..1]; f32 4x16 B: lane holds col, vgpr i -> K = 2*hi+i.
__global__ void repack_win(const float* __restrict__ W, float* __restrict__ B) {
  int t = blockIdx.x * blockDim.x + threadIdx.x;
  const int total = KOFF * 6 * 32 * 2;
  if (t >= total) return;
  int i    = t & 1;
  int lane = (t >> 1) & 31;
  int nb   = (t >> 6) % 6;
  int ko   = t / 384;
  int ci = 2 * (lane >> 4) + i;
  int co = nb * 16 + (lane & 15);
  B[t] = W[((long)ko * 4 + ci) * CH + co];
}

// One wave per (k-offset, 16-entry tile): gather 16 rows of X (f32->f16),
// 3x v_wmma_f32_16x16x32_f16 per 16-col output tile (6 tiles), atomic scatter-add.
__global__ void conv96_wmma(const float* __restrict__ X, const _Float16* __restrict__ Bp,
                            const int* __restrict__ rb_in, const int* __restrict__ rb_out,
                            int P, const int* __restrict__ nin_p, const int* __restrict__ nout_p,
                            float* __restrict__ T) {
  const int lane = threadIdx.x;
  const int hi = lane >> 4, lr = lane & 15;
  const int ko = blockIdx.y;
  const int ebase = blockIdx.x * 16;
  const int n_in = *nin_p, n_out = *nout_p;

  // scatter rows for this lane: D layout lane L holds rows m = v + 8*hi, col = lr.
  // Load these first: if no lane has a live scatter target the whole tile is pad
  // (rulebook entries are front-packed) and the wave exits before any FLOP.
  int orow[8];
  bool live = false;
  #pragma unroll
  for (int v = 0; v < 8; ++v) {
    int ee = ebase + 8 * hi + v;
    int o = (ee < P) ? rb_out[ko * P + ee] : n_out;
    orow[v] = (o < 0 || o >= n_out) ? -1 : o;   // pad entries (== n_out) are dropped
    live |= (orow[v] >= 0);
  }
  if (__builtin_amdgcn_ballot_w32(live) == 0u) return;  // wave-uniform exit

  // gather index for this lane's row (A: lane%16 = row, hi = K-chunk)
  int e = ebase + lr;
  int g = (e < P) ? rb_in[ko * P + e] : n_in;
  g = (g < 0) ? n_in : ((g > n_in) ? n_in : g);   // row n_in is the zero pad row

  v16h a[3];
  const float* rowp = X + (long)g * CH + hi * 8;
  #pragma unroll
  for (int kb = 0; kb < 3; ++kb) {
    const float* p = rowp + kb * 32;
    #pragma unroll
    for (int j = 0; j < 8; ++j) a[kb][j] = (_Float16)p[j];
    #pragma unroll
    for (int j = 0; j < 8; ++j) a[kb][8 + j] = (_Float16)p[16 + j];
  }

  const _Float16* bbase = Bp + (long)ko * 9216;  // 3*6 fragments of 512 halves
  for (int nb = 0; nb < 6; ++nb) {
    v8f c = {};
    #pragma unroll
    for (int kb = 0; kb < 3; ++kb) {
      v16h b = *(const v16h*)(bbase + ((kb * 6 + nb) * 32 + lane) * 16);
      c = __builtin_amdgcn_wmma_f32_16x16x32_f16(false, a[kb], false, b,
                                                 (short)0, c, false, false);
    }
    int col = nb * 16 + lr;
    #pragma unroll
    for (int v = 0; v < 8; ++v)
      if (orow[v] >= 0) atomicAdd(&T[(long)orow[v] * CH + col], c[v]);
  }
}

// Input conv: K = 4 exactly -> V_WMMA_F32_16X16X4_F32 (pure f32 path).
__global__ void convin_wmma(const float* __restrict__ F, const float* __restrict__ Bp,
                            const int* __restrict__ rb_in, const int* __restrict__ rb_out,
                            int P, const int* __restrict__ nin_p, const int* __restrict__ nout_p,
                            float* __restrict__ T) {
  const int lane = threadIdx.x;
  const int hi = lane >> 4, lr = lane & 15;
  const int ko = blockIdx.y;
  const int ebase = blockIdx.x * 16;
  const int n_in = *nin_p, n_out = *nout_p;

  int orow[8];
  bool live = false;
  #pragma unroll
  for (int v = 0; v < 8; ++v) {
    int ee = ebase + 8 * hi + v;
    int o = (ee < P) ? rb_out[ko * P + ee] : n_out;
    orow[v] = (o < 0 || o >= n_out) ? -1 : o;
    live |= (orow[v] >= 0);
  }
  if (__builtin_amdgcn_ballot_w32(live) == 0u) return;  // wave-uniform exit

  int e = ebase + lr;
  int g = (e < P) ? rb_in[ko * P + e] : n_in;
  bool gv = (unsigned)g < (unsigned)n_in;        // features has no pad row
  v2f a;
  a[0] = gv ? F[(long)g * 4 + 2 * hi + 0] : 0.0f;
  a[1] = gv ? F[(long)g * 4 + 2 * hi + 1] : 0.0f;

  const float* bbase = Bp + (long)ko * 384;      // 6 fragments of 64 floats
  for (int nb = 0; nb < 6; ++nb) {
    v2f b = *(const v2f*)(bbase + (nb * 32 + lane) * 2);
    v8f c = {};
    c = __builtin_amdgcn_wmma_f32_16x16x4_f32(false, a, false, b,
                                              (short)0, c, false, false);
    int col = nb * 16 + lr;
    #pragma unroll
    for (int v = 0; v < 8; ++v)
      if (orow[v] >= 0) atomicAdd(&T[(long)orow[v] * CH + col], c[v]);
  }
}

// Per-channel sum / sumsq over the live n rows (n read from device scalar).
__global__ void bn_stats(const float* __restrict__ X, const int* __restrict__ n_p,
                         float* __restrict__ stats) {
  __shared__ float ls[CH], lq[CH];
  int tid = threadIdx.x;
  for (int i = tid; i < CH; i += blockDim.x) { ls[i] = 0.0f; lq[i] = 0.0f; }
  __syncthreads();
  const int n = *n_p;
  const long total = (long)n * CH;
  for (long idx = (long)blockIdx.x * blockDim.x + tid; idx < total;
       idx += (long)gridDim.x * blockDim.x) {
    float v = X[idx];
    int ch = (int)(idx % CH);
    atomicAdd(&ls[ch], v);
    atomicAdd(&lq[ch], v * v);
  }
  __syncthreads();
  for (int i = tid; i < CH; i += blockDim.x) {
    atomicAdd(&stats[i], ls[i]);
    atomicAdd(&stats[CH + i], lq[i]);
  }
}

// y = relu( (x - mean)*rsqrt(var+1e-3)*g + be  [+ resid] ); row n -> zero pad row.
__global__ void bn_apply(const float* __restrict__ T, const float* __restrict__ stats,
                         const float* __restrict__ gamma, const float* __restrict__ beta,
                         const float* __restrict__ resid, float* __restrict__ OUT,
                         const int* __restrict__ n_p, long rows) {
  long idx = (long)blockIdx.x * blockDim.x + threadIdx.x;
  if (idx >= rows * CH) return;
  const int n = *n_p;
  long row = idx / CH;
  int ch = (int)(idx % CH);
  if (row > n) return;
  if (row == n) { OUT[idx] = 0.0f; return; }   // keep zero pad row for gathers
  float inv_n = 1.0f / (float)n;
  float m = stats[ch] * inv_n;
  float var = stats[CH + ch] * inv_n - m * m;
  float y = (T[idx] - m) * rsqrtf(var + 1e-3f) * gamma[ch] + beta[ch];
  if (resid) y += resid[idx];
  OUT[idx] = fmaxf(y, 0.0f);
}

// ---------------- host orchestration ----------------

static void run_zero(float* p, long n, hipStream_t s) {
  long b = (n + 255) / 256; if (b > 4096) b = 4096; if (b < 1) b = 1;
  fill_zero_f32<<<(int)b, 256, 0, s>>>(p, n);
}

static void run_conv96(const float* X, const _Float16* Bp, const int* rin, const int* rout,
                       int P, const int* ninp, const int* noutp, float* T, long boundOut,
                       hipStream_t s) {
  run_zero(T, (boundOut + 1) * CH, s);
  dim3 grid((P + 15) / 16, KOFF);
  conv96_wmma<<<grid, 32, 0, s>>>(X, Bp, rin, rout, P, ninp, noutp, T);
}

static void run_bn(const float* T, const float* g, const float* be, const float* resid,
                   float* out, const int* n_p, long bound, float* stats, hipStream_t s) {
  run_zero(stats, 2 * CH, s);
  long sb = (bound * CH + 255) / 256; if (sb > 512) sb = 512; if (sb < 1) sb = 1;
  bn_stats<<<(int)sb, 256, 0, s>>>(T, n_p, stats);
  long rows = bound + 1;
  bn_apply<<<(int)((rows * CH + 255) / 256), 256, 0, s>>>(T, stats, g, be, resid, out, n_p, rows);
}

extern "C" void kernel_launch(void* const* d_in, const int* in_sizes, int n_in_cnt,
                              void* d_out, int out_size, void* d_ws, size_t ws_size,
                              hipStream_t stream) {
  (void)n_in_cnt; (void)ws_size;
  // d_in layout: [0]=features, [1..59]=params (sorted pytree leaves), [60..69]=rulebooks,
  // [70..72]=n0,n1,n2. Sorted param keys:
  //  1:be_in  2..9:c1{b1,b2,be1,be2,g1,g2,w1,w2}  10..17:c2b  18..25:c3
  //  26..33:c4b  34..41:c5  42..49:c6  50..53:d2{b,be,g,w}  54..57:d4{b,be,g,w}
  //  58:g_in  59:w_in        (biases b* unused: they cancel inside BN)
  const float* feats = (const float*)d_in[0];
  const int n0 = in_sizes[0] / 4;
  const int n2 = out_size / CH;

  auto F = [&](int i) { return (const float*)d_in[i]; };
  auto I = [&](int i) { return (const int*)d_in[i]; };

  const int *rb0_in = I(60),  *rb0_out = I(61);
  const int *rbd1_in = I(62), *rbd1_out = I(63);
  const int *rb1_in = I(64),  *rb1_out = I(65);
  const int *rbd2_in = I(66), *rbd2_out = I(67);
  const int *rb2_in = I(68),  *rb2_out = I(69);
  const int *n0_p = I(70), *n1_p = I(71), *n2_p = I(72);
  const int P0 = in_sizes[60] / 27, Pd1 = in_sizes[62] / 27, P1 = in_sizes[64] / 27;
  const int Pd2 = in_sizes[66] / 27, P2 = in_sizes[68] / 27;

  // workspace carve (256B aligned): 14 f16 weight packs + f32 input pack + 3 act buffers + stats
  char* w = (char*)d_ws;
  auto carve = [&](size_t b) { void* p = (void*)w; w += (b + 255) & ~(size_t)255; return p; };
  const int W96E = KOFF * 3 * 6 * 32 * 16;
  _Float16* wp[14];
  static const int widx[14] = {8, 9, 16, 17, 24, 25, 32, 33, 40, 41, 48, 49, 53, 57};
  for (int i = 0; i < 14; ++i) wp[i] = (_Float16*)carve((size_t)W96E * 2);
  float* wpin = (float*)carve((size_t)KOFF * 6 * 64 * 4);
  const long rows0 = (long)n0 + 1;
  float* X = (float*)carve((size_t)rows0 * CH * 4);
  float* Y = (float*)carve((size_t)rows0 * CH * 4);
  float* T = (float*)carve((size_t)rows0 * CH * 4);
  float* stats = (float*)carve(2 * CH * 4);

  for (int i = 0; i < 14; ++i)
    repack_w96<<<(W96E + 255) / 256, 256, 0, stream>>>(F(widx[i]), wp[i]);
  repack_win<<<(KOFF * 6 * 64 + 255) / 256, 256, 0, stream>>>(F(59), wpin);

  const long b0 = n0;   // host-known upper bound for levels 0 and 1 (n1 <= n0)
  const long b2 = n2;   // exact (out_size/96)

  // input conv 4->96 (f32 WMMA) + BN + ReLU
  run_zero(T, (b0 + 1) * CH, stream);
  { dim3 g((P0 + 15) / 16, KOFF);
    convin_wmma<<<g, 32, 0, stream>>>(feats, wpin, rb0_in, rb0_out, P0, n0_p, n0_p, T); }
  run_bn(T, F(58), F(1), nullptr, X, n0_p, b0, stats, stream);

  auto block = [&](int base, _Float16* w1, _Float16* w2, const int* rin, const int* rout,
                   int P, const int* np, long bound) {
    run_conv96(X, w1, rin, rout, P, np, np, T, bound, stream);
    run_bn(T, F(base + 4), F(base + 2), nullptr, Y, np, bound, stats, stream);   // g1, be1
    run_conv96(Y, w2, rin, rout, P, np, np, T, bound, stream);
    run_bn(T, F(base + 5), F(base + 3), X, Y, np, bound, stats, stream);         // g2, be2, +resid
    float* t = X; X = Y; Y = t;
  };

  block(2, wp[0], wp[1], rb0_in, rb0_out, P0, n0_p, b0);                 // c1
  // d2 downsample (n0 -> n1)
  run_conv96(X, wp[12], rbd1_in, rbd1_out, Pd1, n0_p, n1_p, T, b0, stream);
  run_bn(T, F(52), F(51), nullptr, Y, n1_p, b0, stats, stream);
  { float* t = X; X = Y; Y = t; }
  block(10, wp[2], wp[3], rb1_in, rb1_out, P1, n1_p, b0);                // c2b
  block(18, wp[4], wp[5], rb1_in, rb1_out, P1, n1_p, b0);                // c3
  // d4 downsample (n1 -> n2)
  run_conv96(X, wp[13], rbd2_in, rbd2_out, Pd2, n1_p, n2_p, T, b2, stream);
  run_bn(T, F(56), F(55), nullptr, Y, n2_p, b2, stats, stream);
  { float* t = X; X = Y; Y = t; }
  block(26, wp[6], wp[7], rb2_in, rb2_out, P2, n2_p, b2);                // c4b
  block(34, wp[8], wp[9], rb2_in, rb2_out, P2, n2_p, b2);                // c5
  block(42, wp[10], wp[11], rb2_in, rb2_out, P2, n2_p, b2);              // c6

  copy_f32<<<(out_size + 255) / 256, 256, 0, stream>>>(X, (float*)d_out, out_size);
}
